// UpdateNode_21062519620179
// MI455X (gfx1250) — matrix operand order repaired and checked by reading, not compiled, not run on hardware
//
// CDNA5 / gfx1250 implementation of the MoE-gated equivariant edge update.
//
// Strategy: all channel-mixing matmuls are mapped to v_wmma_f32_16x16x32_f16.
// Weights are pre-swizzled to f16 B-fragment layout (with the 1/sqrt(C) mix
// scales folded in). Each wave owns a 16-edge tile; 8 waves per block share
// B panels staged through LDS (via GLOBAL_LOAD_ASYNC_TO_LDS_B128 when the
// toolchain exposes it). Expert gates are folded into the A-fragment rows.
// Scatter (segment_sum) uses global_atomic_add_f32 into a workspace
// accumulator, finalize kernel applies the residual + one-hot terms.

#include <hip/hip_runtime.h>
#include <hip/hip_bf16.h>
#include <math.h>

typedef __attribute__((ext_vector_type(16))) _Float16 v16h;
typedef __attribute__((ext_vector_type(8)))  float    v8f;
typedef __attribute__((ext_vector_type(4)))  unsigned int u32x4;
typedef __attribute__((__vector_size__(16))) int vsi4;   // matches builtin param int4

union FragH { v16h h; u32x4 q[2]; };

#if defined(__HIP_DEVICE_COMPILE__) && \
    __has_builtin(__builtin_amdgcn_global_load_async_to_lds_b128) && \
    __has_builtin(__builtin_amdgcn_s_wait_asynccnt)
#define USE_ASYNC_LDS 1
#else
#define USE_ASYNC_LDS 0
#endif

// ---- problem dims -----------------------------------------------------------
#define M0C   128
#define M1C   64
#define M2C   32
#define E0C   64
#define C0C   192   // M0+E0
#define C1C   96    // M1+E1
#define C2C   48    // M2+E2
#define T0C   224   // M0+M1+M2
#define LATC  128
#define NEXPC 8
#define NFW   480   // node feature / output width
#define EFW   240   // edge feature width
// ---- fragment tiling --------------------------------------------------------
#define KT0 6
#define NT0 14
#define KT1 3
#define NT1 4
#define KT2 2       // C2=48 zero-padded to 64
#define NT2 2
#define KTL 4       // K tiles for 128-wide inputs (lat, s)
#define NTENV 14
#define NTP0 8
#define KTP1 2
#define NTP1 4
#define KTP2 1
#define NTP2 2

// ---- LDS layout (bytes) -----------------------------------------------------
constexpr int H_X0  = 8 * 16 * 192;   // x0  (reused as s, stride 128)
constexpr int H_X1  = 8 * 48 * 96;    // x1r (3 m-subtiles; reused as v1, stride 64)
constexpr int H_X2  = 8 * 80 * 64;    // x2r (5 m-subtiles, K padded to 64; reused as v2, stride 32)
constexpr int H_LAT = 8 * 16 * 128;
constexpr int H_B   = 14 * 512;       // staged B panel (max NT=14 frags)
constexpr int F_G   = 8 * 16 * 8;     // gates
constexpr int F_D1  = 8 * 16 * 9;
constexpr int F_D2  = 8 * 16 * 25;
constexpr int I_EC  = 8 * 16;
constexpr int I_AE  = 8 * 16;
constexpr int LDS_BYTES =
    (H_X0 + H_X1 + H_X2 + H_LAT + H_B) * 2 + (F_G + F_D1 + F_D2) * 4 + (I_EC + I_AE) * 4;

__device__ __forceinline__ int mini(int a, int b) { return a < b ? a : b; }
__device__ __forceinline__ float sigm(float x) { return 1.f / (1.f + __expf(-x)); }
__device__ __forceinline__ v8f z8() {
  v8f z;
  #pragma unroll
  for (int j = 0; j < 8; ++j) z[j] = 0.f;
  return z;
}

// Build an A fragment (16x32 f16) from row-major f16 LDS.
// ISA layout: lane<16 -> M=lane, halves [0..7]=K 0..7, [8..15]=K 16..23;
//             lane>=16 -> M=lane-16, halves [0..7]=K 8..15, [8..15]=K 24..31.
__device__ __forceinline__ v16h load_a(const _Float16* base, int row, int strideH, int kt, int lane) {
  int c0 = kt * 32 + ((lane & 16) ? 8 : 0);
  const _Float16* p = base + row * strideH + c0;
  FragH f;
  f.q[0] = *(const u32x4*)(p);
  f.q[1] = *(const u32x4*)(p + 16);
  return f.h;
}

__device__ __forceinline__ v16h load_b(const _Float16* ldsB, int nt, int lane) {
  const _Float16* p = ldsB + nt * 512 + lane * 16;
  FragH f;
  f.q[0] = *(const u32x4*)(p);
  f.q[1] = *(const u32x4*)(p + 8);
  return f.h;
}

// Cooperative stage of a B panel (nVec 16B chunks) global -> LDS.
__device__ __forceinline__ void stage_panel(const _Float16* src, _Float16* ldsB, int nVec) {
#if USE_ASYNC_LDS
  const u32x4* gsrc = (const u32x4*)src;
  u32x4* ldst = (u32x4*)ldsB;
  for (int t = threadIdx.x; t < nVec; t += blockDim.x)
    __builtin_amdgcn_global_load_async_to_lds_b128(
        (__attribute__((address_space(1))) vsi4*)(u32x4*)(gsrc + t),
        (__attribute__((address_space(3))) vsi4*)(ldst + t), 0, 0);
  __builtin_amdgcn_s_wait_asynccnt(0);
#else
  for (int t = threadIdx.x; t < nVec; t += blockDim.x)
    ((u32x4*)ldsB)[t] = ((const u32x4*)src)[t];
#endif
}

// Gated multi-expert mix: m[s][nt] += sum_e (g_e * A_s) @ B[e]  (e=0 is shared, g=1).
// All waves in the block call this in lockstep (B panel staged via LDS).
template<int NE, int SUBS, int KT, int NT>
__device__ void mix_phase(const _Float16* xA, int strideH,
                          const _Float16* wsW, int ntTotal, int ntOff,
                          _Float16* ldsB, const float* gates,
                          v8f (&m)[SUBS][NT]) {
  const int lane = threadIdx.x & 31;
  const int er = lane & 15;
  for (int e = 0; e < NE; ++e) {
    _Float16 gh = (_Float16)1.f;
    if (NE > 1 && e > 0) gh = (_Float16)gates[er * 8 + (e - 1)];
    for (int kt = 0; kt < KT; ++kt) {
      // cooperative stage of the B panel (NT frags, 512 halves each)
      const _Float16* src = wsW + ((size_t)(e * KT + kt) * ntTotal + ntOff) * 512;
      stage_panel(src, ldsB, NT * 64);
      __syncthreads();
      v16h a[SUBS];
      #pragma unroll
      for (int s = 0; s < SUBS; ++s) {
        v16h av = load_a(xA, s * 16 + er, strideH, kt, lane);
        if (NE > 1 && e > 0) {
          #pragma unroll
          for (int i = 0; i < 16; ++i) av[i] = av[i] * gh;
        }
        a[s] = av;
      }
      #pragma unroll
      for (int nt = 0; nt < NT; ++nt) {
        v16h b = load_b(ldsB, nt, lane);
        #pragma unroll
        for (int s = 0; s < SUBS; ++s)
          m[s][nt] = __builtin_amdgcn_wmma_f32_16x16x32_f16(
              false, a[s], false, b, (short)0, m[s][nt], false, false);
      }
      __syncthreads();
    }
  }
}

// ---- weight pre-swizzle: fp32 row-major (Cin x Cout) -> f16 B-fragment layout.
// B layout (mirror of A): lane<16 -> N=lane, slots [0..7]=K 0..7, [8..15]=K 16..23; lane>=16 -> N=lane-16, slots [0..7]=K 8..15, [8..15]=K 24..31.
__global__ void prep_weights(const float* __restrict__ Wsh, const float* __restrict__ Wex,
                             int Cin, int Cout, int KT, int NT, float scale,
                             _Float16* __restrict__ dst) {
  int f = blockIdx.x;
  int nt = f % NT; int t1 = f / NT; int kt = t1 % KT; int e = t1 / KT;
  int tid = threadIdx.x;               // 0..511, one half per thread
  int lane = tid >> 4, slot = tid & 15;
  int k = kt * 32 + ((lane & 16) >> 1) + (slot & 7) + ((slot & 8) << 1);
  int n = nt * 16 + (lane & 15);
  float v = 0.f;
  if (k < Cin && n < Cout) {
    const float* src = (e == 0) ? Wsh : (Wex + (size_t)(e - 1) * Cin * Cout);
    v = src[(size_t)k * Cout + n] * scale;
  }
  dst[(size_t)f * 512 + tid] = (_Float16)v;
}

__global__ void zero_acc(float* acc, int n) {
  int i = blockIdx.x * blockDim.x + threadIdx.x;
  if (i < n) acc[i] = 0.f;
}

__global__ void prep_mgwm(const float* __restrict__ mg, const float* __restrict__ Wm,
                          float* __restrict__ out) {
  int x = threadIdx.x;
  if (x < NEXPC) {
    float s = 0.f;
    for (int c = 0; c < LATC; ++c) s += mg[c] * Wm[c * NEXPC + x];
    out[x] = s;
  }
}

// ---- main edge kernel -------------------------------------------------------
__global__ __launch_bounds__(256) void edge_kernel(
    const float* __restrict__ latents, const float* __restrict__ node_features,
    const float* __restrict__ edge_features,
    const float* __restrict__ wigner_D1, const float* __restrict__ wigner_D2,
    const float* __restrict__ Wg, const float* __restrict__ mgWm,
    const int* __restrict__ edge_index, const int* __restrict__ active_edges,
    const float* __restrict__ bp0,
    const _Float16* __restrict__ wsW0, const _Float16* __restrict__ wsW1,
    const _Float16* __restrict__ wsW2, const _Float16* __restrict__ wsWenv,
    const _Float16* __restrict__ wsWp0, const _Float16* __restrict__ wsWp1,
    const _Float16* __restrict__ wsWp2,
    float* __restrict__ acc, int Etot) {
  extern __shared__ char smem[];
  _Float16* sX0  = (_Float16*)smem;
  _Float16* sX1  = sX0 + H_X0;
  _Float16* sX2  = sX1 + H_X1;
  _Float16* sLAT = sX2 + H_X2;
  _Float16* sB   = sLAT + H_LAT;
  float* sG  = (float*)(sB + H_B);
  float* sD1 = sG + F_G;
  float* sD2 = sD1 + F_D1;
  int* sEC = (int*)(sD2 + F_D2);
  int* sAE = sEC + I_EC;

  const int wave = threadIdx.x >> 5;
  const int lane = threadIdx.x & 31;
  const int er = lane & 15;
  const int tileBase = (blockIdx.x * 8 + wave) * 16;

  _Float16* x0w  = sX0 + wave * (16 * 192);
  _Float16* x1w  = sX1 + wave * (48 * 96);
  _Float16* x2w  = sX2 + wave * (80 * 64);
  _Float16* latw = sLAT + wave * (16 * 128);
  float* gtw = sG + wave * 128;
  float* d1w = sD1 + wave * 144;
  float* d2w = sD2 + wave * 400;
  int* ecw = sEC + wave * 16;
  int* aew = sAE + wave * 16;

  // ---- prologue: gather, rotate, gates -------------------------------------
  if (lane < 16) {
    int eid = mini(tileBase + lane, Etot - 1);
    int ae = active_edges[eid];
    aew[lane] = ae;
    ecw[lane] = edge_index[ae];       // row 0 of (2,E)
  }
  float* sWg = (float*)sB;            // stage Wg (128x8 f32) in the B buffer for now
  for (int t = threadIdx.x; t < LATC * NEXPC; t += blockDim.x) sWg[t] = Wg[t];
  __syncthreads();

  for (int t = lane; t < 16 * LATC; t += 32) {   // latents -> f16 LDS
    int i = t >> 7, c = t & 127;
    float v = latents[(size_t)aew[i] * LATC + c];
    latw[t] = (_Float16)(((tileBase + i) < Etot) ? v : 0.f);
  }
  for (int t = lane; t < 16 * 9; t += 32) {
    int eid = mini(tileBase + t / 9, Etot - 1);
    d1w[t] = wigner_D1[(size_t)eid * 9 + (t % 9)];
  }
  for (int t = lane; t < 16 * 25; t += 32) {
    int eid = mini(tileBase + t / 25, Etot - 1);
    d2w[t] = wigner_D2[(size_t)eid * 25 + (t % 25)];
  }
  for (int t = lane; t < 16 * C0C; t += 32) {    // x0 = [n0 | e0]
    int i = t / C0C, c = t % C0C;
    int eid = mini(tileBase + i, Etot - 1);
    float v = (c < M0C) ? node_features[(size_t)ecw[i] * NFW + c]
                        : edge_features[(size_t)eid * EFW + (c - M0C)];
    x0w[i * C0C + c] = (_Float16)(((tileBase + i) < Etot) ? v : 0.f);
  }
  __syncthreads();                               // D1/D2 visible across lanes

  for (int t = lane; t < 16 * C1C; t += 32) {    // x1 rotate: x1r[c][m] = sum_n D1[m][n] x1[c][n]
    int i = t / C1C, c = t % C1C;
    int eid = mini(tileBase + i, Etot - 1);
    bool valid = (tileBase + i) < Etot;
    float xv[3];
    #pragma unroll
    for (int n = 0; n < 3; ++n)
      xv[n] = (c < M1C) ? node_features[(size_t)ecw[i] * NFW + M0C + c * 3 + n]
                        : edge_features[(size_t)eid * EFW + E0C + (c - M1C) * 3 + n];
    #pragma unroll
    for (int mm = 0; mm < 3; ++mm) {
      float r = d1w[i * 9 + mm * 3 + 0] * xv[0] + d1w[i * 9 + mm * 3 + 1] * xv[1] +
                d1w[i * 9 + mm * 3 + 2] * xv[2];
      x1w[(mm * 16 + i) * C1C + c] = (_Float16)(valid ? r : 0.f);
    }
  }
  for (int t = lane; t < 16 * C2C; t += 32) {    // x2 rotate (5x5)
    int i = t / C2C, c = t % C2C;
    int eid = mini(tileBase + i, Etot - 1);
    bool valid = (tileBase + i) < Etot;
    float xv[5];
    #pragma unroll
    for (int n = 0; n < 5; ++n)
      xv[n] = (c < M2C) ? node_features[(size_t)ecw[i] * NFW + M0C + 3 * M1C + c * 5 + n]
                        : edge_features[(size_t)eid * EFW + E0C + 3 * 32 + (c - M2C) * 5 + n];
    #pragma unroll
    for (int mm = 0; mm < 5; ++mm) {
      float r = 0.f;
      #pragma unroll
      for (int n = 0; n < 5; ++n) r += d2w[i * 25 + mm * 5 + n] * xv[n];
      x2w[(mm * 16 + i) * 64 + c] = (_Float16)(valid ? r : 0.f);
    }
  }
  for (int t = lane; t < 80 * 16; t += 32)       // zero-pad x2 K 48..63
    x2w[(t >> 4) * 64 + 48 + (t & 15)] = (_Float16)0.f;
  __syncthreads();

  for (int t = lane; t < 16 * NEXPC; t += 32) {  // gate logits
    int i = t >> 3, xg = t & 7;
    float s = mgWm[xg];
    for (int c = 0; c < LATC; ++c) s += (float)latw[i * LATC + c] * sWg[c * NEXPC + xg];
    gtw[i * NEXPC + xg] = s;
  }
  __syncthreads();
  if (lane < 16) {                               // softmax per edge
    float* g = gtw + lane * NEXPC;
    if ((tileBase + lane) < Etot) {
      float mx = g[0];
      for (int x = 1; x < 8; ++x) mx = fmaxf(mx, g[x]);
      float s = 0.f;
      for (int x = 0; x < 8; ++x) { float e_ = __expf(g[x] - mx); g[x] = e_; s += e_; }
      float inv = 1.f / s;
      for (int x = 0; x < 8; ++x) g[x] *= inv;
    } else {
      for (int x = 0; x < 8; ++x) g[x] = 0.f;
    }
  }
  __syncthreads();

  // ---- phases --------------------------------------------------------------
  v8f g1[4], g2[2];
  {
    v8f m0[1][NT0];
    #pragma unroll
    for (int nt = 0; nt < NT0; ++nt) m0[0][nt] = z8();
    mix_phase<1 + NEXPC, 1, KT0, NT0>(x0w, C0C, wsW0, NT0, 0, sB, gtw, m0);
    // activations: tiles 0..7 -> silu(s) staged to LDS; 8..13 -> sigmoid gates
    #pragma unroll
    for (int nt = 0; nt < 4; ++nt) {
      #pragma unroll
      for (int r = 0; r < 8; ++r) g1[nt][r] = sigm(m0[0][8 + nt][r]);
    }
    #pragma unroll
    for (int nt = 0; nt < 2; ++nt) {
      #pragma unroll
      for (int r = 0; r < 8; ++r) g2[nt][r] = sigm(m0[0][12 + nt][r]);
    }
    _Float16* sS = x0w;                          // reuse, stride 128
    #pragma unroll
    for (int nt = 0; nt < 8; ++nt) {
      int col = nt * 16 + er;
      #pragma unroll
      for (int r = 0; r < 8; ++r) {
        int em = r + ((lane & 16) ? 8 : 0);
        float tv = m0[0][nt][r];
        sS[em * 128 + col] = (_Float16)(tv * sigm(tv));
      }
    }
  }
  {
    v8f mp0[1][NTP0], mw[1][NTP0];
    #pragma unroll
    for (int nt = 0; nt < NTP0; ++nt) { mp0[0][nt] = z8(); mw[0][nt] = z8(); }
    mix_phase<1, 1, KTL, NTP0>(x0w, 128, wsWp0, NTP0, 0, sB, gtw, mp0);
    mix_phase<1, 1, KTL, NTP0>(latw, 128, wsWenv, NTENV, 0, sB, gtw, mw);
    #pragma unroll
    for (int nt = 0; nt < NTP0; ++nt) {          // epilogue l=0 scatter
      int col = nt * 16 + er;
      float bp = bp0[col];
      #pragma unroll
      for (int r = 0; r < 8; ++r) {
        int em = r + ((lane & 16) ? 8 : 0);
        if ((tileBase + em) < Etot)
          atomicAdd(&acc[(size_t)ecw[em] * NFW + col], (mp0[0][nt][r] + bp) * mw[0][nt][r]);
      }
    }
  }
  {
    v8f m1[3][NT1];
    #pragma unroll
    for (int s = 0; s < 3; ++s) {
      #pragma unroll
      for (int nt = 0; nt < NT1; ++nt) m1[s][nt] = z8();
    }
    mix_phase<1 + NEXPC, 3, KT1, NT1>(x1w, C1C, wsW1, NT1, 0, sB, gtw, m1);
    _Float16* sV1 = x1w;                         // reuse, stride 64, rows = n*16+edge
    #pragma unroll
    for (int nt = 0; nt < NT1; ++nt) {           // rotate back + sigmoid gate
      int col = nt * 16 + er;
      #pragma unroll
      for (int r = 0; r < 8; ++r) {
        int em = r + ((lane & 16) ? 8 : 0);
        float gv = g1[nt][r];
        float a0 = m1[0][nt][r], a1 = m1[1][nt][r], a2 = m1[2][nt][r];
        #pragma unroll
        for (int n = 0; n < 3; ++n) {
          float rot = d1w[em * 9 + n * 3 + 0] * a0 + d1w[em * 9 + n * 3 + 1] * a1 +
                      d1w[em * 9 + n * 3 + 2] * a2;
          sV1[(n * 16 + em) * 64 + col] = (_Float16)(rot * gv);
        }
      }
    }
  }
  {
    v8f mp1[3][NTP1], mw1[1][NTP1];
    #pragma unroll
    for (int s = 0; s < 3; ++s) {
      #pragma unroll
      for (int nt = 0; nt < NTP1; ++nt) mp1[s][nt] = z8();
    }
    #pragma unroll
    for (int nt = 0; nt < NTP1; ++nt) mw1[0][nt] = z8();
    mix_phase<1, 3, KTP1, NTP1>(x1w, 64, wsWp1, NTP1, 0, sB, gtw, mp1);
    mix_phase<1, 1, KTL, NTP1>(latw, 128, wsWenv, NTENV, 8, sB, gtw, mw1);
    #pragma unroll
    for (int nt = 0; nt < NTP1; ++nt) {          // epilogue l=1 scatter
      int col = nt * 16 + er;
      #pragma unroll
      for (int r = 0; r < 8; ++r) {
        int em = r + ((lane & 16) ? 8 : 0);
        if ((tileBase + em) < Etot) {
          size_t base = (size_t)ecw[em] * NFW + M0C + col * 3;
          #pragma unroll
          for (int n = 0; n < 3; ++n)
            atomicAdd(&acc[base + n], mp1[n][nt][r] * mw1[0][nt][r]);
        }
      }
    }
  }
  {
    v8f m2[5][NT2];
    #pragma unroll
    for (int s = 0; s < 5; ++s) {
      #pragma unroll
      for (int nt = 0; nt < NT2; ++nt) m2[s][nt] = z8();
    }
    mix_phase<1 + NEXPC, 5, KT2, NT2>(x2w, 64, wsW2, NT2, 0, sB, gtw, m2);
    _Float16* sV2 = x2w;                         // reuse, stride 32
    #pragma unroll
    for (int nt = 0; nt < NT2; ++nt) {
      int col = nt * 16 + er;
      #pragma unroll
      for (int r = 0; r < 8; ++r) {
        int em = r + ((lane & 16) ? 8 : 0);
        float gv = g2[nt][r];
        float am[5];
        #pragma unroll
        for (int s = 0; s < 5; ++s) am[s] = m2[s][nt][r];
        #pragma unroll
        for (int n = 0; n < 5; ++n) {
          float rot = 0.f;
          #pragma unroll
          for (int mm = 0; mm < 5; ++mm) rot += d2w[em * 25 + n * 5 + mm] * am[mm];
          sV2[(n * 16 + em) * 32 + col] = (_Float16)(rot * gv);
        }
      }
    }
  }
  {
    v8f mp2[5][NTP2], mw2[1][NTP2];
    #pragma unroll
    for (int s = 0; s < 5; ++s) {
      #pragma unroll
      for (int nt = 0; nt < NTP2; ++nt) mp2[s][nt] = z8();
    }
    #pragma unroll
    for (int nt = 0; nt < NTP2; ++nt) mw2[0][nt] = z8();
    mix_phase<1, 5, KTP2, NTP2>(x2w, 32, wsWp2, NTP2, 0, sB, gtw, mp2);
    mix_phase<1, 1, KTL, NTP2>(latw, 128, wsWenv, NTENV, 12, sB, gtw, mw2);
    #pragma unroll
    for (int nt = 0; nt < NTP2; ++nt) {          // epilogue l=2 scatter
      int col = nt * 16 + er;
      #pragma unroll
      for (int r = 0; r < 8; ++r) {
        int em = r + ((lane & 16) ? 8 : 0);
        if ((tileBase + em) < Etot) {
          size_t base = (size_t)ecw[em] * NFW + M0C + 3 * M1C + col * 5;
          #pragma unroll
          for (int n = 0; n < 5; ++n)
            atomicAdd(&acc[base + n], mp2[n][nt][r] * mw2[0][nt][r]);
        }
      }
    }
  }
}

// ---- finalize: residual + one-hot term -------------------------------------
__global__ void finalize(const float* __restrict__ acc, const float* __restrict__ nf,
                         const float* __restrict__ onehot, const float* __restrict__ Woh0,
                         const float* __restrict__ Woh1, const float* __restrict__ Woh2,
                         float* __restrict__ out, int N) {
  const float NORM = 0.22360679775f;       // 1/sqrt(20)
  const float C_OLD = 0.894427190999916f;  // 1/sqrt(1.25)
  const float C_NEW = 0.447213595499958f;
  const float INVS = 0.102597835208515f;   // 1/sqrt(95)
  int idx = blockIdx.x * blockDim.x + threadIdx.x;
  if (idx >= N * NFW) return;
  int i = idx / NFW, col = idx % NFW;
  float h = C_OLD * nf[idx] + C_NEW * NORM * acc[idx];
  const float* W;
  int ch;
  if (col < M0C) { ch = col; W = Woh0; }
  else if (col < M0C + 3 * M1C) { ch = (col - M0C) / 3; W = Woh1; }
  else { ch = (col - M0C - 3 * M1C) / 5; W = Woh2; }
  const float* oh = onehot + (size_t)i * 95;
  const float* w = W + (size_t)ch * 95;
  float dot = 0.f;
  for (int t = 0; t < 95; ++t) dot += oh[t] * w[t];
  out[idx] = h + h * dot * INVS;
}

// ---- host side --------------------------------------------------------------
extern "C" void kernel_launch(void* const* d_in, const int* in_sizes, int n_in,
                              void* d_out, int out_size, void* d_ws, size_t ws_size,
                              hipStream_t stream) {
  const float* latents       = (const float*)d_in[0];
  const float* node_features = (const float*)d_in[1];
  const float* edge_features = (const float*)d_in[2];
  const float* mole_globals  = (const float*)d_in[4];
  const float* wigner_D1     = (const float*)d_in[5];
  const float* wigner_D2     = (const float*)d_in[6];
  const float* node_onehot   = (const float*)d_in[7];
  const float* Wg   = (const float*)d_in[8];
  const float* Wm   = (const float*)d_in[9];
  const float* W0_e = (const float*)d_in[10];
  const float* W0_s = (const float*)d_in[11];
  const float* W1_e = (const float*)d_in[12];
  const float* W1_s = (const float*)d_in[13];
  const float* W2_e = (const float*)d_in[14];
  const float* W2_s = (const float*)d_in[15];
  const float* Wp0  = (const float*)d_in[16];
  const float* bp0  = (const float*)d_in[17];
  const float* Wp1  = (const float*)d_in[18];
  const float* Wp2  = (const float*)d_in[19];
  const float* Wenv = (const float*)d_in[20];
  const float* Woh0 = (const float*)d_in[21];
  const float* Woh1 = (const float*)d_in[22];
  const float* Woh2 = (const float*)d_in[23];
  const int* edge_index   = (const int*)d_in[25];
  const int* active_edges = (const int*)d_in[26];

  int E = in_sizes[0] / LATC;
  int N = in_sizes[1] / NFW;

  char* ws = (char*)d_ws;
  size_t off = 0;
  auto carve = [&](size_t bytes) -> char* {
    char* p = ws + off; off = (off + bytes + 255) & ~(size_t)255; return p;
  };
  float* acc = (float*)carve((size_t)N * NFW * sizeof(float));
  float* mgWm = (float*)carve(64);
  _Float16* wsW0   = (_Float16*)carve((size_t)9 * KT0 * NT0 * 512 * 2);
  _Float16* wsW1   = (_Float16*)carve((size_t)9 * KT1 * NT1 * 512 * 2);
  _Float16* wsW2   = (_Float16*)carve((size_t)9 * KT2 * NT2 * 512 * 2);
  _Float16* wsWenv = (_Float16*)carve((size_t)KTL * NTENV * 512 * 2);
  _Float16* wsWp0  = (_Float16*)carve((size_t)KTL * NTP0 * 512 * 2);
  _Float16* wsWp1  = (_Float16*)carve((size_t)KTP1 * NTP1 * 512 * 2);
  _Float16* wsWp2  = (_Float16*)carve((size_t)KTP2 * NTP2 * 512 * 2);

  (void)hipFuncSetAttribute((const void*)edge_kernel,
                            hipFuncAttributeMaxDynamicSharedMemorySize, LDS_BYTES);

  int nElem = N * NFW;
  zero_acc<<<(nElem + 255) / 256, 256, 0, stream>>>(acc, nElem);
  prep_mgwm<<<1, 64, 0, stream>>>(mole_globals, Wm, mgWm);

  prep_weights<<<9 * KT0 * NT0, 512, 0, stream>>>(W0_s, W0_e, C0C, T0C, KT0, NT0,
                                                  0.0721687836487032f, wsW0);
  prep_weights<<<9 * KT1 * NT1, 512, 0, stream>>>(W1_s, W1_e, C1C, M1C, KT1, NT1,
                                                  0.102062072615966f, wsW1);
  prep_weights<<<9 * KT2 * NT2, 512, 0, stream>>>(W2_s, W2_e, C2C, M2C, KT2, NT2,
                                                  0.144337567297406f, wsW2);
  prep_weights<<<KTL * NTENV, 512, 0, stream>>>(Wenv, nullptr, LATC, T0C, KTL, NTENV,
                                                0.0883883476483184f, wsWenv);
  prep_weights<<<KTL * NTP0, 512, 0, stream>>>(Wp0, nullptr, M0C, M0C, KTL, NTP0,
                                               0.0883883476483184f, wsWp0);
  prep_weights<<<KTP1 * NTP1, 512, 0, stream>>>(Wp1, nullptr, M1C, M1C, KTP1, NTP1,
                                                0.125f, wsWp1);
  prep_weights<<<KTP2 * NTP2, 512, 0, stream>>>(Wp2, nullptr, M2C, M2C, KTP2, NTP2,
                                                0.176776695296637f, wsWp2);

  int nBlocks = (E + 127) / 128;   // 8 waves x 16 edges per block
  edge_kernel<<<nBlocks, 256, LDS_BYTES, stream>>>(
      latents, node_features, edge_features, wigner_D1, wigner_D2, Wg, mgWm,
      edge_index, active_edges, bp0, wsW0, wsW1, wsW2, wsWenv, wsWp0, wsWp1, wsWp2,
      acc, E);

  finalize<<<(nElem + 255) / 256, 256, 0, stream>>>(acc, node_features, node_onehot,
                                                    Woh0, Woh1, Woh2, (float*)d_out, N);
}
